// GroupedQueryAttention_86105504350233
// MI455X (gfx1250) — matrix operand order repaired and checked by reading, not compiled, not run on hardware
//
#include <hip/hip_runtime.h>
#include <hip/hip_bf16.h>

typedef _Float16 v8h  __attribute__((ext_vector_type(8)));
typedef _Float16 v16h __attribute__((ext_vector_type(16)));
typedef float    v8f  __attribute__((ext_vector_type(8)));

#define B_  2
#define S_  2048
#define D_  2048
#define H_  32
#define KV_ 8
#define HD_ 64

// ---------------------------------------------------------------------------
// A-fragment loader for v_wmma_f32_16x16x32_f16.
// Lane holds row M=lane%16; in this code's own k order element e maps to
//   K = 16*(lane/16) + e   -> one contiguous 32B run (two 16B chunks).
// ---------------------------------------------------------------------------
static __device__ __forceinline__ v16h load_a_frag(const _Float16* rowptr, int kk, int half) {
  v8h lo = *(const v8h*)(rowptr + kk + half * 16);
  v8h hi = *(const v8h*)(rowptr + kk + half * 16 + 8);
  v16h a;
#pragma unroll
  for (int i = 0; i < 8; i++) { a[i] = lo[i]; a[8 + i] = hi[i]; }
  return a;
}

// ---------------------------------------------------------------------------
// Elementwise f32 -> f16 convert
// ---------------------------------------------------------------------------
__global__ __launch_bounds__(256) void cvt_f16_kernel(const float* __restrict__ in,
                                                      _Float16* __restrict__ out, long n) {
  long i = (long)blockIdx.x * blockDim.x + threadIdx.x;
  if (i < n) out[i] = (_Float16)in[i];
}

// ---------------------------------------------------------------------------
// Transpose + convert: in[K][N] f32 -> out[N][K] f16
// ---------------------------------------------------------------------------
__global__ __launch_bounds__(256) void transpose_f16_kernel(const float* __restrict__ in,
                                                            _Float16* __restrict__ out,
                                                            int K, int N) {
  long i = (long)blockIdx.x * blockDim.x + threadIdx.x;
  if (i >= (long)K * N) return;
  int k = (int)(i / N);
  int n = (int)(i % N);
  out[(long)n * K + k] = (_Float16)in[i];
}

// ---------------------------------------------------------------------------
// WMMA GEMM: C[M][N] = A[M][K] * Bt[N][K]^T   (f16 inputs, f32 accumulate)
// 32x64 tile per wave. Load phase (12 b128 into distinct regs) then compute
// phase (8 WMMAs) so the scheduler can overlap loads with matrix ops instead
// of s_wait_loadcnt 0 before every WMMA.
// ---------------------------------------------------------------------------
template <int F16OUT>
__global__ __launch_bounds__(256) void gemm_wmma_kernel(const _Float16* __restrict__ A,
                                                        const _Float16* __restrict__ Bt,
                                                        void* __restrict__ C,
                                                        int M, int N, int K) {
  const int lane = threadIdx.x & 31;
  const int warp = threadIdx.x >> 5;
  const int half = lane >> 4;
  const int lm   = lane & 15;
  const int nt64 = N >> 6;
  long task = (long)blockIdx.x * 8 + warp;
  int mt = (int)(task / nt64);     // 32-row tiles
  int nb = (int)(task % nt64);     // 64-col tiles
  if (mt >= (M >> 5)) return;
  const int m0 = mt << 5;
  const int n0 = nb << 6;

  const _Float16* arow0 = A + (long)(m0 + lm) * K;
  const _Float16* arow1 = A + (long)(m0 + 16 + lm) * K;
  v8f acc[2][4] = {};
  for (int kk = 0; kk < K; kk += 32) {
    // ---- load phase: all fragments into distinct registers ----
    v16h af0 = load_a_frag(arow0, kk, half);
    v16h af1 = load_a_frag(arow1, kk, half);
    v16h bf[4];
#pragma unroll
    for (int t = 0; t < 4; t++)
      bf[t] = *(const v16h*)(Bt + (long)(n0 + t * 16 + lm) * K + kk + half * 16);
    if (kk + 32 < K) {  // gfx1250 global_prefetch_b8 for next A K-chunk
      __builtin_prefetch(arow0 + kk + 32, 0, 1);
      __builtin_prefetch(arow1 + kk + 32, 0, 1);
    }
    // ---- compute phase: 8 WMMAs ----
#pragma unroll
    for (int t = 0; t < 4; t++) {
      acc[0][t] = __builtin_amdgcn_wmma_f32_16x16x32_f16(false, af0, false, bf[t],
                                                         (short)0, acc[0][t], false, false);
      acc[1][t] = __builtin_amdgcn_wmma_f32_16x16x32_f16(false, af1, false, bf[t],
                                                         (short)0, acc[1][t], false, false);
    }
  }
  // C/D layout: VGPR r, half h -> row = r + 8*h, col = lane%16
#pragma unroll
  for (int mi = 0; mi < 2; mi++)
#pragma unroll
    for (int t = 0; t < 4; t++)
#pragma unroll
      for (int r = 0; r < 8; r++) {
        long row = m0 + mi * 16 + r + half * 8;
        long col = n0 + t * 16 + lm;
        if (F16OUT) ((_Float16*)C)[row * N + col] = (_Float16)acc[mi][t][r];
        else        ((float*)C)[row * N + col]    = acc[mi][t][r];
      }
}

// ---------------------------------------------------------------------------
// RoPE + layout rearrange.
// in : [B, S, NH*64] f16 (GEMM output)
// mode 0 (Q/K): apply RoPE, write [B, NH, S, 64]
// mode 1 (V)  : no RoPE,    write [B, NH, 64, S]  (transposed for P*V B-frags)
// ---------------------------------------------------------------------------
__global__ __launch_bounds__(256) void rope_rearrange_kernel(const _Float16* __restrict__ in,
                                                             _Float16* __restrict__ out,
                                                             int NH, int mode) {
  long i = (long)blockIdx.x * blockDim.x + threadIdx.x;
  long total = (long)B_ * S_ * NH * HD_;
  if (i >= total) return;
  int d = (int)(i & 63);
  long t = i >> 6;
  int head = (int)(t % NH); t /= NH;
  int s = (int)(t % S_);
  int b = (int)(t / S_);
  const _Float16* row = in + ((long)b * S_ + s) * (NH * HD_) + head * HD_;
  float v = (float)row[d];
  if (mode == 0) {
    int fi = d & 31;  // freq index = d % (HD/2)
    // inv_freq = 10000^(-fi/32) = exp(-fi * ln(10000)/32)
    float invf = __expf(-(float)fi * (9.210340371976184f / 32.0f));
    float ang = (float)s * invf;
    float c, sn;
    sincosf(ang, &sn, &c);
    float partner = (float)row[(d < 32) ? (d + 32) : (d - 32)];
    v = (d < 32) ? (v * c - partner * sn) : (v * c + partner * sn);
    out[(((long)b * NH + head) * S_ + s) * HD_ + d] = (_Float16)v;
  } else {
    out[(((long)b * NH + head) * HD_ + d) * S_ + s] = (_Float16)v;
  }
}

// ---------------------------------------------------------------------------
// Flash attention, causal, grouped KV. One wave per (b, h, 16-row q tile).
// 64-wide KV chunks: 8 score WMMAs + 2 rowsum WMMAs (ones trick) + 8 PV WMMAs.
// K fragments preloaded before the score phase; V fragments issued right after
// so they fly during softmax + LDS staging and are consumed by the PV WMMAs.
// Qh : [B, H, S, 64]  Kh : [B, KV, S, 64]  Vth : [B, KV, 64, S]   (all f16)
// Oh : [B*S, H*64] f16
// ---------------------------------------------------------------------------
__global__ __launch_bounds__(32) void attn_wmma_kernel(const _Float16* __restrict__ Qh,
                                                       const _Float16* __restrict__ Kh,
                                                       const _Float16* __restrict__ Vth,
                                                       _Float16* __restrict__ Oh) {
  __shared__ _Float16 pbuf[16 * 64];  // P tile staging: C-layout -> A-fragment shuffle
  const int lane = threadIdx.x & 31;
  const int half = lane >> 4;
  const int lm   = lane & 15;
  const int qt = blockIdx.x;        // S/16 tiles
  const int bh = blockIdx.y;        // B*H
  const int b  = bh >> 5;           // H = 32
  const int h  = bh & 31;
  const int kv = h >> 2;            // REP = 4
  const int q0 = qt * 16;

  const _Float16* qbase = Qh  + ((long)(b * H_  + h ) * S_ + q0) * HD_;
  const _Float16* kbase = Kh  + ((long)(b * KV_ + kv) * S_) * HD_;
  const _Float16* vbase = Vth + ((long)(b * KV_ + kv) * HD_) * S_;

  // Q fragments for d=[0,32) and d=[32,64)
  const _Float16* qrow = qbase + lm * HD_;
  v16h qa[2];
#pragma unroll
  for (int w = 0; w < 2; w++) {
    v8h lo = *(const v8h*)(qrow + w * 32 + half * 16);
    v8h hi = *(const v8h*)(qrow + w * 32 + half * 16 + 8);
#pragma unroll
    for (int i = 0; i < 8; i++) { qa[w][i] = lo[i]; qa[w][8 + i] = hi[i]; }
  }

  // All-ones B fragment: P x ones gives row sums in every output column
  v16h vones;
#pragma unroll
  for (int i = 0; i < 16; i++) vones[i] = (_Float16)1.0f;

  float mrun[8], lrun[8];
#pragma unroll
  for (int r = 0; r < 8; r++) { mrun[r] = -1e30f; lrun[r] = 0.0f; }
  v8f o[4] = {};

  const int jmax = (q0 + 15) >> 6;  // inclusive 64-wide KV chunk index (causal)
  for (int jt = 0; jt <= jmax; jt++) {
    const int j0 = jt * 64;
    // ---- K load phase: 8 fragments (16 b128) into distinct registers ----
    v16h kbf[4][2];
#pragma unroll
    for (int st = 0; st < 4; st++) {
      const _Float16* krow = kbase + (long)(j0 + st * 16 + lm) * HD_;
      kbf[st][0] = *(const v16h*)(krow + 0  + half * 16);  // d in [0,32)
      kbf[st][1] = *(const v16h*)(krow + 32 + half * 16);  // d in [32,64)
    }
    // ---- score phase: 4 tiles x (2 WMMAs over d) ----
    v8f sc[4];
#pragma unroll
    for (int st = 0; st < 4; st++) {
      v8f s = {};
      s = __builtin_amdgcn_wmma_f32_16x16x32_f16(false, qa[0], false, kbf[st][0], (short)0, s, false, false);
      s = __builtin_amdgcn_wmma_f32_16x16x32_f16(false, qa[1], false, kbf[st][1], (short)0, s, false, false);
      sc[st] = s;
    }
    // ---- V load phase: issue now, consumed only by the PV WMMAs below ----
    v16h vbf[4][2];
#pragma unroll
    for (int t = 0; t < 4; t++) {
      const _Float16* vrow = vbase + (long)(t * 16 + lm) * S_ + j0;
      vbf[t][0] = *(const v16h*)(vrow + 0  + half * 16);
      vbf[t][1] = *(const v16h*)(vrow + 32 + half * 16);
    }
    // ---- scale + causal mask ----
#pragma unroll
    for (int st = 0; st < 4; st++)
#pragma unroll
      for (int r = 0; r < 8; r++) {
        int qi = q0 + r + half * 8;
        int kj = j0 + st * 16 + lm;
        sc[st][r] = (kj <= qi) ? sc[st][r] * 0.125f : -1e30f;  // scale = 1/sqrt(64)
      }
    // ---- online max (only max needs cross-lane shuffles) ----
    float alpha[8];
#pragma unroll
    for (int r = 0; r < 8; r++) {
      float t = fmaxf(fmaxf(sc[0][r], sc[1][r]), fmaxf(sc[2][r], sc[3][r]));
      for (int off = 1; off < 16; off <<= 1) t = fmaxf(t, __shfl_xor(t, off));
      float nm = fmaxf(mrun[r], t);
      alpha[r] = __expf(mrun[r] - nm);
      mrun[r] = nm;
#pragma unroll
      for (int st = 0; st < 4; st++) sc[st][r] = __expf(sc[st][r] - nm);
    }
    // ---- stage P (16x64, C-layout) to LDS, re-read as two A-fragments ----
#pragma unroll
    for (int st = 0; st < 4; st++)
#pragma unroll
      for (int r = 0; r < 8; r++)
        pbuf[(r + half * 8) * 64 + st * 16 + lm] = (_Float16)sc[st][r];
    asm volatile("s_wait_dscnt 0" ::: "memory");
    v16h pa[2];
    {
      const _Float16* pr = pbuf + lm * 64;
#pragma unroll
      for (int w = 0; w < 2; w++)
#pragma unroll
        for (int g = 0; g < 2; g++)
#pragma unroll
          for (int i = 0; i < 8; i++)
            pa[w][g * 8 + i] = pr[w * 32 + half * 16 + g * 8 + i];
    }
    // ---- row sums via WMMA with all-ones B (no shuffles) ----
    v8f lsum = {};
    lsum = __builtin_amdgcn_wmma_f32_16x16x32_f16(false, pa[0], false, vones, (short)0, lsum, false, false);
    lsum = __builtin_amdgcn_wmma_f32_16x16x32_f16(false, pa[1], false, vones, (short)0, lsum, false, false);
#pragma unroll
    for (int r = 0; r < 8; r++) lrun[r] = lrun[r] * alpha[r] + lsum[r];
#pragma unroll
    for (int t = 0; t < 4; t++)
#pragma unroll
      for (int r = 0; r < 8; r++) o[t][r] *= alpha[r];
    // ---- O += P * V  (4 output d-tiles x 2 k-chunks) ----
#pragma unroll
    for (int t = 0; t < 4; t++) {
      o[t] = __builtin_amdgcn_wmma_f32_16x16x32_f16(false, pa[0], false, vbf[t][0], (short)0, o[t], false, false);
      o[t] = __builtin_amdgcn_wmma_f32_16x16x32_f16(false, pa[1], false, vbf[t][1], (short)0, o[t], false, false);
    }
  }

  // Normalize and write Oh[b*S + qi][h*64 + d] as f16
#pragma unroll
  for (int t = 0; t < 4; t++)
#pragma unroll
    for (int r = 0; r < 8; r++) {
      long row = (long)b * S_ + q0 + r + half * 8;
      long col = h * HD_ + t * 16 + lm;
      Oh[row * (H_ * HD_) + col] = (_Float16)(o[t][r] / lrun[r]);
    }
}

// ---------------------------------------------------------------------------
extern "C" void kernel_launch(void* const* d_in, const int* in_sizes, int n_in,
                              void* d_out, int out_size, void* d_ws, size_t ws_size,
                              hipStream_t stream) {
  const float* x  = (const float*)d_in[0];
  const float* Wq = (const float*)d_in[1];
  const float* Wk = (const float*)d_in[2];
  const float* Wv = (const float*)d_in[3];
  const float* Wo = (const float*)d_in[4];
  float* out = (float*)d_out;

  const size_t MB = (size_t)1 << 20;
  if (ws_size < 76 * MB) return;
  char* ws = (char*)d_ws;
  _Float16* xh   = (_Float16*)(ws + 0);        // 16 MB  [4096][2048]
  _Float16* Wqt  = (_Float16*)(ws + 16 * MB);  //  8 MB  [2048][2048]
  _Float16* Wkt  = (_Float16*)(ws + 24 * MB);  //  2 MB  [512][2048]
  _Float16* Wvt  = (_Float16*)(ws + 26 * MB);  //  2 MB  [512][2048]
  _Float16* Wot  = (_Float16*)(ws + 28 * MB);  //  8 MB  [2048][2048]
  _Float16* Qtmp = (_Float16*)(ws + 36 * MB);  // 16 MB  [4096][2048]
  _Float16* Ktmp = (_Float16*)(ws + 52 * MB);  //  4 MB  [4096][512]
  _Float16* Vtmp = (_Float16*)(ws + 56 * MB);  //  4 MB  [4096][512]
  _Float16* Qh   = (_Float16*)(ws + 60 * MB);  // 16 MB  [B,H,S,64]
  // Reuse dead regions (ordering on one stream guarantees safety):
  _Float16* Kh  = Wkt;  // 4 MB over Wkt+Wvt (dead after K/V GEMMs)
  _Float16* Vth = Wqt;  // 4 MB over Wqt      (dead after Q GEMM)
  _Float16* Oh  = xh;   // 16 MB over xh      (dead after projections)

  const int M = B_ * S_;  // 4096

  // 1) x -> f16
  long nx = (long)M * D_;
  cvt_f16_kernel<<<(int)((nx + 255) / 256), 256, 0, stream>>>(x, xh, nx);

  // 2) transpose + convert weights
  transpose_f16_kernel<<<(int)(((long)D_ * 2048 + 255) / 256), 256, 0, stream>>>(Wq, Wqt, D_, 2048);
  transpose_f16_kernel<<<(int)(((long)D_ * 512 + 255) / 256), 256, 0, stream>>>(Wk, Wkt, D_, 512);
  transpose_f16_kernel<<<(int)(((long)D_ * 512 + 255) / 256), 256, 0, stream>>>(Wv, Wvt, D_, 512);
  transpose_f16_kernel<<<(int)(((long)2048 * 2048 + 255) / 256), 256, 0, stream>>>(Wo, Wot, 2048, 2048);

  // 3) QKV projections (WMMA, 32x64 tiles per wave)
  {
    long tasks = (long)(M / 32) * (2048 / 64);
    gemm_wmma_kernel<1><<<(int)((tasks + 7) / 8), 256, 0, stream>>>(xh, Wqt, Qtmp, M, 2048, D_);
  }
  {
    long tasks = (long)(M / 32) * (512 / 64);
    gemm_wmma_kernel<1><<<(int)((tasks + 7) / 8), 256, 0, stream>>>(xh, Wkt, Ktmp, M, 512, D_);
    gemm_wmma_kernel<1><<<(int)((tasks + 7) / 8), 256, 0, stream>>>(xh, Wvt, Vtmp, M, 512, D_);
  }

  // 4) RoPE + rearrange
  {
    long nq = (long)B_ * S_ * H_ * HD_;
    rope_rearrange_kernel<<<(int)((nq + 255) / 256), 256, 0, stream>>>(Qtmp, Qh, H_, 0);
    long nk = (long)B_ * S_ * KV_ * HD_;
    rope_rearrange_kernel<<<(int)((nk + 255) / 256), 256, 0, stream>>>(Ktmp, Kh, KV_, 0);
    rope_rearrange_kernel<<<(int)((nk + 255) / 256), 256, 0, stream>>>(Vtmp, Vth, KV_, 1);
  }

  // 5) Flash attention (WMMA)
  {
    dim3 grid(S_ / 16, B_ * H_);
    attn_wmma_kernel<<<grid, 32, 0, stream>>>(Qh, Kh, Vth, Oh);
  }

  // 6) Output projection (WMMA, f32 out)
  {
    long tasks = (long)(M / 32) * (2048 / 64);
    gemm_wmma_kernel<0><<<(int)((tasks + 7) / 8), 256, 0, stream>>>(Oh, Wot, out, M, 2048, H_ * HD_);
  }
}
